// MLPdirected_55628416417859
// MI455X (gfx1250) — matrix operand order repaired and checked by reading, not compiled, not run on hardware
//
#include <hip/hip_runtime.h>
#include <hip/hip_bf16.h>
#include <cmath>

// Problem constants (B, F, N, H from the reference)
#define Bn 8
#define Fn 32
#define Nn 512
#define Hn 8

typedef __attribute__((ext_vector_type(16))) _Float16 v16h;
typedef __attribute__((ext_vector_type(8)))  float    v8f;

// ---------------------------------------------------------------------------
// Kernel 1: sw[b*N+j] = (sum_k adj[0,k,j] - adj[b,j,j]) / (N-1)
// Coalesced: for each k, the 512 threads read a contiguous row of adj[0].
// ---------------------------------------------------------------------------
__global__ void k_sw(const float* __restrict__ adj, float* __restrict__ sw) {
    const int j = blockIdx.x * blockDim.x + threadIdx.x;
    if (j >= Nn) return;
    float acc = 0.f;
#pragma unroll 8
    for (int k = 0; k < Nn; ++k) acc += adj[k * Nn + j];
    const float inv = 1.0f / (float)(Nn - 1);
#pragma unroll
    for (int b = 0; b < Bn; ++b) {
        const float diag = adj[(size_t)b * Nn * Nn + (size_t)j * Nn + j];
        sw[b * Nn + j] = (acc - diag) * inv;
    }
}

// ---------------------------------------------------------------------------
// Kernel 2: one wave per 16-row tile of the flattened [B*N, 32] "et" matrix.
//   Y[16x16] = et_tile[16x32] @ [W1a | W1c][32x16]   via v_wmma_f32_16x16x32_f16
//   cols 0..7  -> Aout[r,h]
//   cols 8..15 -> Cout[r,h] = Y + sw[r]*W1[64,h] + b1[h]
// et[r=b*N+i, f] = emb[b*F*N + f*N + i]  (emb is [B,F,N], already "transposed")
// Writeback is fully branchless: pointer + fixup selected per lane once.
// ---------------------------------------------------------------------------
__global__ void k_gemm(const float* __restrict__ emb, const float* __restrict__ W1,
                       const float* __restrict__ b1,  const float* __restrict__ sw,
                       float* __restrict__ Aout, float* __restrict__ Cout) {
    const int lane = threadIdx.x;        // 0..31, one wave32 per block
    const int r0   = blockIdx.x * 16;    // row tile in flattened [B*N]
    const int b    = r0 / Nn;
    const int i0   = r0 % Nn;            // tile never straddles batches (16 | 512)
    const int M    = lane & 15;
    const int half = lane >> 4;

    const float* embB = emb + (size_t)b * Fn * Nn;

    // A fragment (16x32 f16): lane holds row M; elems 0..7 = K {0..7}+8*half,
    //                         elems 8..15 = K {16..23}+8*half
    v16h afrag;
#pragma unroll
    for (int p = 0; p < 8; ++p) {
        const int k0 = p + 8 * half;
        const int k1 = 16 + p + 8 * half;
        afrag[p]     = (_Float16)embB[k0 * Nn + i0 + M];
        afrag[p + 8] = (_Float16)embB[k1 * Nn + i0 + M];
    }

    // B fragment (32x16 f16): column ncol = lane&15; element p = K = (p&7) + 8*half + 16*(p>>3).
    // Columns 0..7 <- W1 rows 0..31 (src block); 8..15 <- W1 rows 32..63 (dst).
    const int ncol = lane & 15;
    v16h bfrag;
#pragma unroll
    for (int p = 0; p < 16; ++p) {
        const int k = (p & 7) + 8 * half + ((p >> 3) << 4);
        const float w = (ncol < Hn) ? W1[k * Hn + ncol]
                                    : W1[(32 + k) * Hn + (ncol - Hn)];
        bfrag[p] = (_Float16)w;
    }

    v8f cacc = {};
    // (neg_a, A, neg_b, B, c_mod, C, reuse_a, reuse_b)
    v8f d = __builtin_amdgcn_wmma_f32_16x16x32_f16(
        false, afrag, false, bfrag, (short)0, cacc, false, false);

    // ---- Branchless writeback ----
    // A-columns (ncol<8): dst = Aout + ncol, fixup = 0.
    // C-columns (ncol>=8): dst = Cout + ncol-8, fixup = sw[row]*W1[64,h] + b1[h].
    const bool  isC = (ncol >= Hn);
    const int   h   = isC ? (ncol - Hn) : 0;              // clamped, safe
    const float sel = isC ? 1.0f : 0.0f;
    const float fscale = W1[64 * Hn + h] * sel;
    const float fbias  = b1[h] * sel;
    float* dst = (isC ? (Cout + (ncol - Hn)) : (Aout + ncol));

    // sw for this lane's 8 consecutive rows: two aligned float4 loads
    const int rbase = r0 + 8 * half;
    const float4* swp = reinterpret_cast<const float4*>(sw + rbase);
    const float4 s0 = swp[0];
    const float4 s1 = swp[1];
    float swv[8] = { s0.x, s0.y, s0.z, s0.w, s1.x, s1.y, s1.z, s1.w };

#pragma unroll
    for (int m = 0; m < 8; ++m) {
        const int row = rbase + m;
        dst[(size_t)row * Hn] = fmaf(swv[m], fscale, d[m] + fbias);
    }
}

// ---------------------------------------------------------------------------
// Kernel 3: streaming pass, HBM-write-bound.
// Block = 256 threads handles a (16 i) x (512 j) tile of one batch:
//   out[b,i,j] = sigmoid(b2 + sum_h W2[h] * relu(A[b,i,h] + C[b,j,h]))
// A tile lives in LDS (broadcast reads); C reads and output stores coalesced.
// ---------------------------------------------------------------------------
__global__ void k_out(const float* __restrict__ Aout, const float* __restrict__ Cout,
                      const float* __restrict__ W2,  const float* __restrict__ b2,
                      float* __restrict__ out) {
    __shared__ float At[16][Hn];
    const int blk = blockIdx.x;
    const int b   = blk >> 5;            // / (N/16)
    const int i0  = (blk & 31) << 4;     // * 16
    const int tid = threadIdx.x;

    if (tid < 16 * Hn) {
        const int ii = tid >> 3, h = tid & 7;
        At[ii][h] = Aout[((size_t)(b * Nn + i0) + ii) * Hn + h];
    }
    float w2r[Hn];
#pragma unroll
    for (int h = 0; h < Hn; ++h) w2r[h] = W2[h];
    const float bias2 = b2[0];
    __syncthreads();

    for (int jj = tid; jj < Nn; jj += 256) {
        if (jj + 256 < Nn) {
            // gfx1250 global_prefetch_b8 for the next C chunk
            __builtin_prefetch(&Cout[((size_t)(b * Nn + jj + 256)) * Hn], 0, 1);
        }
        const float* crow = &Cout[((size_t)(b * Nn + jj)) * Hn];
        float c8[Hn];
#pragma unroll
        for (int h = 0; h < Hn; ++h) c8[h] = crow[h];

#pragma unroll
        for (int ii = 0; ii < 16; ++ii) {
            float acc = bias2;
#pragma unroll
            for (int h = 0; h < Hn; ++h) {
                const float v = At[ii][h] + c8[h];
                acc = fmaf(w2r[h], fmaxf(v, 0.0f), acc);
            }
            const float s = 1.0f / (1.0f + __expf(-acc));
            out[(size_t)b * Nn * Nn + (size_t)(i0 + ii) * Nn + jj] = s;
        }
    }
}

// ---------------------------------------------------------------------------
// Inputs (setup_inputs order): adj_in, emb_in, W1, b1, W2, b2
// Workspace layout: sw[B*N] | Aout[B*N*H] | Cout[B*N*H]  (~272 KB)
// ---------------------------------------------------------------------------
extern "C" void kernel_launch(void* const* d_in, const int* in_sizes, int n_in,
                              void* d_out, int out_size, void* d_ws, size_t ws_size,
                              hipStream_t stream) {
    const float* adj = (const float*)d_in[0];
    const float* emb = (const float*)d_in[1];
    const float* W1  = (const float*)d_in[2];
    const float* b1  = (const float*)d_in[3];
    const float* W2  = (const float*)d_in[4];
    const float* b2  = (const float*)d_in[5];
    float* out = (float*)d_out;

    float* sw   = (float*)d_ws;              // B*N
    float* Aout = sw + Bn * Nn;              // B*N*H
    float* Cout = Aout + (size_t)Bn * Nn * Hn;

    // 1) scaled column-sum / diagonal term
    k_sw<<<Nn / 256, 256, 0, stream>>>(adj, sw);

    // 2) fused [A|C] GEMM via WMMA: one wave per 16-row tile, 256 tiles
    k_gemm<<<(Bn * Nn) / 16, 32, 0, stream>>>(emb, W1, b1, sw, Aout, Cout);

    // 3) streaming outer pass: 256 blocks x 256 threads, writes 8 MB
    k_out<<<Bn * (Nn / 16), 256, 0, stream>>>(Aout, Cout, W2, b2, out);
}
